// TauEndToEndSimple_51144470561437
// MI455X (gfx1250) — compile-verified, matrix operand order checked by reading, not run on hardware
//
#include <hip/hip_runtime.h>
#include <cstdint>
#include <cstddef>

typedef __bf16 bf16;
typedef __attribute__((ext_vector_type(16))) __bf16 v16bf;
typedef __attribute__((ext_vector_type(8)))  float  v8f;

#define NPF   131072   // particles
#define NJ    8192     // jets
#define DIN   36       // pf input features
#define DEMB  512
#define DJF   2056     // 8 + 4*512
#define DJFP  2080     // padded to multiple of 32
#define K0PAD 64       // padded K for emb layer 0
#define MT    32       // rows (M) per block

// ---------------------------------------------------------------------------
// helpers
// ---------------------------------------------------------------------------
__device__ __forceinline__ float eluf(float x) { return x > 0.f ? x : expm1f(x); }

// A/B fragment load for v_wmma_f32_16x16x32_bf16.
// lane = (m = lane&15, khalf = lane>>4); element e -> K = (e&7) + (e>=8?16:0) + 8*khalf
__device__ __forceinline__ v16bf load_frag(const bf16* rowptr, int kc, int khalf) {
  const bf16* p = rowptr + kc * 32 + khalf * 8;
  v16bf a;
#pragma unroll
  for (int i = 0; i < 8; ++i) { a[i] = p[i]; a[8 + i] = p[16 + i]; }
  return a;
}

__device__ __forceinline__ v8f wmma_bf16(v16bf a, v16bf b, v8f c) {
  return __builtin_amdgcn_wmma_f32_16x16x32_bf16(false, a, false, b, (short)0, c,
                                                 false, false);
}

// raw cross-lane xor-add using a precomputed ds_bpermute byte-address
__device__ __forceinline__ float bperm_add(float v, int pa) {
  return v + __int_as_float(__builtin_amdgcn_ds_bpermute(pa, __float_as_int(v)));
}

// float atomic max via signed/unsigned integer atomics (init with -inf bits)
__device__ __forceinline__ void atomicMaxFloat(float* addr, float v) {
  if (v >= 0.f) atomicMax((int*)addr, __float_as_int(v));
  else          atomicMin((unsigned int*)addr, __float_as_uint(v));
}

// ---------------------------------------------------------------------------
// GEMM sweep: block = 32 rows x 512 cols. 8 waves; each wave: 2 row-tiles x
// 4 col-tiles (8 v8f accumulators). B-fragment loaded once per (kc, nt) and
// reused across both row-tiles; A-fragments once per (kc, rowt).
// ---------------------------------------------------------------------------
__device__ __forceinline__ void gemm_tiles(
    const bf16* a0row, const bf16* a1row,   // lane's activation rows (m, 16+m)
    const bf16* Wt, int kstride, int kcount,
    v8f (&acc)[2][4], int (&ncol)[4], int wave, int m, int khalf) {
  const v8f z = {0.f, 0.f, 0.f, 0.f, 0.f, 0.f, 0.f, 0.f};
  const bf16* wrow[4];
#pragma unroll
  for (int nt = 0; nt < 4; ++nt) {
    ncol[nt] = (wave + nt * 8) * 16 + m;
    wrow[nt] = Wt + (size_t)ncol[nt] * kstride;
    acc[0][nt] = z; acc[1][nt] = z;
  }
#pragma unroll 2
  for (int kc = 0; kc < kcount; ++kc) {
    v16bf a0 = load_frag(a0row, kc, khalf);
    v16bf a1 = load_frag(a1row, kc, khalf);
#pragma unroll
    for (int nt = 0; nt < 4; ++nt) {
      v16bf b = load_frag(wrow[nt], kc, khalf);
      acc[0][nt] = wmma_bf16(a0, b, acc[0][nt]);
      acc[1][nt] = wmma_bf16(a1, b, acc[1][nt]);
    }
  }
}

// bias + LayerNorm + ELU computed from accumulators (no f32 staging buffer):
// wave32 bpermute reduction -> per-(row,wave) LDS partials -> 32-thread finish,
// then normalize in registers and store bf16 activations.
__device__ __forceinline__ void ln_epilogue(
    v8f (&acc)[2][4], const int (&ncol)[4],
    const float* bias, const float* g, const float* be,
    bf16 (&Xs)[MT][DEMB],
    float (&redsum)[MT][8], float (&redsq)[MT][8],
    float (&mu_s)[MT], float (&rs_s)[MT],
    int tid, int wave, int lane, int m, int khalf) {
  float bv[4], gv[4], bev[4];
#pragma unroll
  for (int nt = 0; nt < 4; ++nt) {
    bv[nt] = bias[ncol[nt]]; gv[nt] = g[ncol[nt]]; bev[nt] = be[ncol[nt]];
  }
#pragma unroll
  for (int rt = 0; rt < 2; ++rt)
#pragma unroll
    for (int nt = 0; nt < 4; ++nt)
#pragma unroll
      for (int r = 0; r < 8; ++r) acc[rt][nt][r] += bv[nt];

  // precomputed bpermute byte-addresses for xor distances 1,2,4,8 (in-range on wave32)
  int pa[4];
#pragma unroll
  for (int i = 0; i < 4; ++i) pa[i] = (lane ^ (1 << i)) << 2;

  // per-row partial sums over this wave's 64 columns (4 nt x 16 lanes)
#pragma unroll
  for (int rt = 0; rt < 2; ++rt)
#pragma unroll
    for (int r = 0; r < 8; ++r) {
      float s  = acc[rt][0][r] + acc[rt][1][r] + acc[rt][2][r] + acc[rt][3][r];
      float s2 = acc[rt][0][r] * acc[rt][0][r] + acc[rt][1][r] * acc[rt][1][r] +
                 acc[rt][2][r] * acc[rt][2][r] + acc[rt][3][r] * acc[rt][3][r];
#pragma unroll
      for (int i = 0; i < 4; ++i) {
        s  = bperm_add(s, pa[i]);
        s2 = bperm_add(s2, pa[i]);
      }
      if (m == 0) {
        int row = rt * 16 + khalf * 8 + r;
        redsum[row][wave] = s;
        redsq[row][wave]  = s2;
      }
    }
  __syncthreads();   // also fences this layer's Xs reads vs. writes below
  if (tid < MT) {
    float ts = 0.f, ts2 = 0.f;
#pragma unroll
    for (int w = 0; w < 8; ++w) { ts += redsum[tid][w]; ts2 += redsq[tid][w]; }
    float mu = ts * (1.f / DEMB);
    float var = ts2 * (1.f / DEMB) - mu * mu;
    mu_s[tid] = mu;
    rs_s[tid] = rsqrtf(var + 1e-5f);
  }
  __syncthreads();
#pragma unroll
  for (int rt = 0; rt < 2; ++rt)
#pragma unroll
    for (int r = 0; r < 8; ++r) {
      int row = rt * 16 + khalf * 8 + r;
      float mu = mu_s[row], rs = rs_s[row];
#pragma unroll
      for (int nt = 0; nt < 4; ++nt) {
        float v = (acc[rt][nt][r] - mu) * rs * gv[nt] + bev[nt];
        Xs[row][ncol[nt]] = (bf16)eluf(v);
      }
    }
  __syncthreads();
}

// ---------------------------------------------------------------------------
// weight convert: W[k][n] f32  ->  Wt[n][Kpad] bf16 (zero-padded K)
// ---------------------------------------------------------------------------
__global__ void cvt_wt_kernel(const float* __restrict__ W, bf16* __restrict__ Wt,
                              int K, int N, int Kpad) {
  size_t gid = (size_t)blockIdx.x * blockDim.x + threadIdx.x;
  size_t total = (size_t)N * Kpad;
  if (gid >= total) return;
  int n = (int)(gid / Kpad);
  int k = (int)(gid - (size_t)n * Kpad);
  Wt[gid] = (k < K) ? (bf16)W[(size_t)k * N + n] : (bf16)0.f;
}

__global__ void fill_kernel(float* __restrict__ p, float v, size_t n) {
  size_t gid = (size_t)blockIdx.x * blockDim.x + threadIdx.x;
  if (gid < n) p[gid] = v;
}

// ---------------------------------------------------------------------------
// embedding FFN: 36 -> 512 (x5 LN+ELU) -> 512, outer ELU, store bf16
// ---------------------------------------------------------------------------
struct EmbArgs {
  const float* pf;      // [NPF][36]
  const bf16* W0t;      // [512][64]
  const bf16* Wt[5];    // layers 1..5 [512][512]
  const float* b[6];
  const float* g[5];
  const float* be[5];
  bf16* out;            // [NPF][512]
};

__global__ __launch_bounds__(256) void emb_ffn_kernel(EmbArgs A) {
  __shared__ bf16 Xs[MT][DEMB];     // 32 KB
  __shared__ bf16 X0[MT][K0PAD];    // 4 KB
  __shared__ float redsum[MT][8], redsq[MT][8];
  __shared__ float mu_s[MT], rs_s[MT];

  int tid = threadIdx.x;
  int lane = tid & 31, wave = tid >> 5;
  int m = lane & 15, khalf = lane >> 4;
  int rowbase = blockIdx.x * MT;

  // cooperative stage of the 32x36 input tile into zero-padded 32x64 bf16 LDS
  for (int i = tid; i < MT * K0PAD; i += 256) {
    int r = i >> 6, k = i & (K0PAD - 1);
    float v = (k < DIN) ? A.pf[(size_t)(rowbase + r) * DIN + k] : 0.f;
    X0[r][k] = (bf16)v;
  }
  __syncthreads();

  v8f acc[2][4];
  int ncol[4];

  // ---- layer 0 (K padded to 64)
  gemm_tiles(&X0[m][0], &X0[16 + m][0], A.W0t, K0PAD, K0PAD / 32, acc, ncol, wave, m, khalf);
  ln_epilogue(acc, ncol, A.b[0], A.g[0], A.be[0], Xs, redsum, redsq, mu_s, rs_s,
              tid, wave, lane, m, khalf);

  // ---- layers 1..4
  for (int L = 1; L <= 4; ++L) {
    gemm_tiles(&Xs[m][0], &Xs[16 + m][0], A.Wt[L - 1], DEMB, DEMB / 32, acc, ncol, wave, m, khalf);
    ln_epilogue(acc, ncol, A.b[L], A.g[L], A.be[L], Xs, redsum, redsq, mu_s, rs_s,
                tid, wave, lane, m, khalf);
  }

  // ---- layer 5: bias + outer ELU, straight from registers to global bf16
  gemm_tiles(&Xs[m][0], &Xs[16 + m][0], A.Wt[4], DEMB, DEMB / 32, acc, ncol, wave, m, khalf);
#pragma unroll
  for (int nt = 0; nt < 4; ++nt) {
    float bv = A.b[5][ncol[nt]];
#pragma unroll
    for (int rt = 0; rt < 2; ++rt)
#pragma unroll
      for (int r = 0; r < 8; ++r) {
        int row = rt * 16 + khalf * 8 + r;
        float v = acc[rt][nt][r] + bv;
        A.out[(size_t)(rowbase + row) * DEMB + ncol[nt]] = (bf16)eluf(v);
      }
  }
}

// ---------------------------------------------------------------------------
// aggregation passes (segment reductions over sorted jet ids)
// ---------------------------------------------------------------------------
__global__ void agg_pass1(const bf16* __restrict__ x, const int* __restrict__ idx,
                          const float* __restrict__ tptr,
                          float* __restrict__ sum, float* __restrict__ sumsq,
                          float* __restrict__ maxx, float* __restrict__ zmax,
                          float* __restrict__ cnt) {
  size_t gid = (size_t)blockIdx.x * blockDim.x + threadIdx.x;
  size_t total = (size_t)NPF * DEMB;
  if (gid >= total) return;
  int n = (int)(gid >> 9);
  int c = (int)(gid & 511);
  int j = idx[n];
  float xv = (float)x[gid];
  float t = *tptr;
  size_t o = (size_t)j * DEMB + c;
  atomicAdd(&sum[o], xv);
  atomicAdd(&sumsq[o], xv * xv);
  atomicMaxFloat(&maxx[o], xv);
  atomicMaxFloat(&zmax[o], xv * t);
  if (c == 0) atomicAdd(&cnt[j], 1.0f);
}

__global__ void agg_pass2(const bf16* __restrict__ x, const int* __restrict__ idx,
                          const float* __restrict__ tptr,
                          const float* __restrict__ zmax,
                          float* __restrict__ den, float* __restrict__ num) {
  size_t gid = (size_t)blockIdx.x * blockDim.x + threadIdx.x;
  size_t total = (size_t)NPF * DEMB;
  if (gid >= total) return;
  int n = (int)(gid >> 9);
  int c = (int)(gid & 511);
  int j = idx[n];
  float xv = (float)x[gid];
  float t = *tptr;
  size_t o = (size_t)j * DEMB + c;
  float mval = zmax[o];
  if (!isfinite(mval)) mval = 0.f;
  float e = expf(xv * t - mval);
  atomicAdd(&den[o], e);
  atomicAdd(&num[o], xv * e);
}

// build padded bf16 jet feature matrix [NJ][DJFP]
__global__ void finalize_kernel(const float* __restrict__ jetf,
                                const float* __restrict__ sum,
                                const float* __restrict__ sumsq,
                                const float* __restrict__ maxx,
                                const float* __restrict__ den,
                                const float* __restrict__ num,
                                const float* __restrict__ cnt,
                                bf16* __restrict__ feats) {
  int j = blockIdx.x;
  int tid = threadIdx.x;
  float cj = cnt[j];
  float safe = fmaxf(cj, 1.f);
  bf16* row = feats + (size_t)j * DJFP;
  if (tid < 8) row[tid] = (bf16)jetf[(size_t)j * 8 + tid];
  if (tid >= 8 && tid < 32) row[2048 + tid] = (bf16)0.f;  // pad 2056..2079
  for (int c = tid; c < DEMB; c += 256) {
    size_t o = (size_t)j * DEMB + c;
    float mean = sum[o] / safe;
    float mx = (cj > 0.f) ? maxx[o] : 0.f;
    float mean2 = sumsq[o] / safe;
    float var = mean2 - mean * mean;
    float stdv = sqrtf(fmaxf(var, 0.f) + 1e-5f);
    float sm = (cj > 0.f) ? num[o] / den[o] : 0.f;
    row[8 + c]        = (bf16)eluf(mean);
    row[8 + 512 + c]  = (bf16)eluf(mx);
    row[8 + 1024 + c] = (bf16)eluf(stdv);
    row[8 + 1536 + c] = (bf16)eluf(sm);
  }
}

// ---------------------------------------------------------------------------
// head FFN: 2056(pad 2080) -> 512 (x5 LN+ELU) -> nout (2 or 4)
// ---------------------------------------------------------------------------
struct HeadArgs {
  const bf16* feats;    // [NJ][2080]
  const bf16* W0t;      // [512][2080]
  const bf16* Wt[4];    // layers 1..4 [512][512]
  const float* b[5];
  const float* g[5];
  const float* be[5];
  const float* W5;      // [512][nout] raw f32
  const float* b5;      // [nout]
  const float* jetf;    // p4 head: multiply by jet_features[:, :4]; else null
  float* out;           // [NJ][nout]
  int nout;
};

__global__ __launch_bounds__(256) void head_ffn_kernel(HeadArgs A) {
  __shared__ bf16 Xs[MT][DEMB];
  __shared__ float redsum[MT][8], redsq[MT][8];
  __shared__ float mu_s[MT], rs_s[MT];

  int tid = threadIdx.x;
  int lane = tid & 31, wave = tid >> 5;
  int m = lane & 15, khalf = lane >> 4;
  int rowbase = blockIdx.x * MT;

  v8f acc[2][4];
  int ncol[4];

  // ---- layer 0: K=2080, A streamed from global bf16
  gemm_tiles(A.feats + (size_t)(rowbase + m) * DJFP,
             A.feats + (size_t)(rowbase + 16 + m) * DJFP,
             A.W0t, DJFP, DJFP / 32, acc, ncol, wave, m, khalf);
  ln_epilogue(acc, ncol, A.b[0], A.g[0], A.be[0], Xs, redsum, redsq, mu_s, rs_s,
              tid, wave, lane, m, khalf);

  // ---- layers 1..4
  for (int L = 1; L <= 4; ++L) {
    gemm_tiles(&Xs[m][0], &Xs[16 + m][0], A.Wt[L - 1], DEMB, DEMB / 32, acc, ncol, wave, m, khalf);
    ln_epilogue(acc, ncol, A.b[L], A.g[L], A.be[L], Xs, redsum, redsq, mu_s, rs_s,
                tid, wave, lane, m, khalf);
  }

  // ---- final small linear 512 -> nout (+ optional p4 scaling)
  if (tid < MT * A.nout) {
    int r = tid / A.nout, o = tid - r * A.nout;
    float acc5 = A.b5[o];
    const bf16* h = &Xs[r][0];
    for (int k = 0; k < DEMB; ++k) acc5 += (float)h[k] * A.W5[(size_t)k * A.nout + o];
    int j = rowbase + r;
    if (A.jetf) acc5 *= A.jetf[(size_t)j * 8 + o];
    A.out[(size_t)j * A.nout + o] = acc5;
  }
}

// ---------------------------------------------------------------------------
// host side
// ---------------------------------------------------------------------------
struct FFNP { const float* W[6]; const float* b[6]; const float* g[5]; const float* be[5]; };

static int parse_ffn(void* const* d_in, int p, FFNP& f, bool sorted_keys) {
  for (int i = 0; i < 6; ++i) f.W[i] = (const float*)d_in[p++];
  for (int i = 0; i < 6; ++i) f.b[i] = (const float*)d_in[p++];
  if (sorted_keys) {  // jax pytree: W, b, beta, g
    for (int i = 0; i < 5; ++i) f.be[i] = (const float*)d_in[p++];
    for (int i = 0; i < 5; ++i) f.g[i]  = (const float*)d_in[p++];
  } else {            // insertion order: W, b, g, beta
    for (int i = 0; i < 5; ++i) f.g[i]  = (const float*)d_in[p++];
    for (int i = 0; i < 5; ++i) f.be[i] = (const float*)d_in[p++];
  }
  return p;
}

extern "C" void kernel_launch(void* const* d_in, const int* in_sizes, int n_in,
                              void* d_out, int out_size, void* d_ws, size_t ws_size,
                              hipStream_t stream) {
  // --- resolve input pointer order ---
  const float *jetf = nullptr, *pf = nullptr, *tptr = nullptr;
  const int* idx = nullptr;
  FFNP emb, istau, p4;
  bool sorted_keys = (in_sizes[0] == DIN * DEMB);  // emb W0 first => sorted pytree
  if (sorted_keys) {
    int p = 0;
    p = parse_ffn(d_in, p, emb, true);
    p = parse_ffn(d_in, p, istau, true);
    jetf = (const float*)d_in[p++];
    pf   = (const float*)d_in[p++];
    idx  = (const int*)d_in[p++];
    p = parse_ffn(d_in, p, p4, true);
    tptr = (const float*)d_in[p++];
  } else {
    int p = 0;
    jetf = (const float*)d_in[p++];
    pf   = (const float*)d_in[p++];
    idx  = (const int*)d_in[p++];
    p = parse_ffn(d_in, p, emb, false);
    p = parse_ffn(d_in, p, istau, false);
    p = parse_ffn(d_in, p, p4, false);
    tptr = (const float*)d_in[p++];
  }

  // --- workspace carve-out ---
  char* ws = (char*)d_ws;
  auto carve = [&](size_t bytes) {
    void* r = (void*)ws;
    ws += (bytes + 255) & ~(size_t)255;
    return r;
  };
  bf16* pf_enc = (bf16*)carve((size_t)NPF * DEMB * sizeof(bf16));
  bf16* feats  = (bf16*)carve((size_t)NJ * DJFP * sizeof(bf16));
  float* sum   = (float*)carve((size_t)NJ * DEMB * sizeof(float));
  float* sumsq = (float*)carve((size_t)NJ * DEMB * sizeof(float));
  float* maxx  = (float*)carve((size_t)NJ * DEMB * sizeof(float));
  float* zmax  = (float*)carve((size_t)NJ * DEMB * sizeof(float));
  float* den   = (float*)carve((size_t)NJ * DEMB * sizeof(float));
  float* num   = (float*)carve((size_t)NJ * DEMB * sizeof(float));
  float* cnt   = (float*)carve((size_t)NJ * sizeof(float));
  bf16* embW0t = (bf16*)carve((size_t)DEMB * K0PAD * sizeof(bf16));
  bf16* embWt  = (bf16*)carve((size_t)5 * DEMB * DEMB * sizeof(bf16));
  bf16* hW0t[2]; bf16* hWt[2];
  for (int h = 0; h < 2; ++h) {
    hW0t[h] = (bf16*)carve((size_t)DEMB * DJFP * sizeof(bf16));
    hWt[h]  = (bf16*)carve((size_t)4 * DEMB * DEMB * sizeof(bf16));
  }

  const int T = 256;
  auto nblk = [](size_t total, int t) { return (unsigned)((total + t - 1) / t); };

  // --- convert weights to transposed K-padded bf16 ---
  cvt_wt_kernel<<<nblk((size_t)DEMB * K0PAD, T), T, 0, stream>>>(emb.W[0], embW0t, DIN, DEMB, K0PAD);
  for (int i = 1; i <= 5; ++i)
    cvt_wt_kernel<<<nblk((size_t)DEMB * DEMB, T), T, 0, stream>>>(
        emb.W[i], embWt + (size_t)(i - 1) * DEMB * DEMB, DEMB, DEMB, DEMB);
  FFNP* heads[2] = {&istau, &p4};
  for (int h = 0; h < 2; ++h) {
    cvt_wt_kernel<<<nblk((size_t)DEMB * DJFP, T), T, 0, stream>>>(heads[h]->W[0], hW0t[h], DJF, DEMB, DJFP);
    for (int i = 1; i <= 4; ++i)
      cvt_wt_kernel<<<nblk((size_t)DEMB * DEMB, T), T, 0, stream>>>(
          heads[h]->W[i], hWt[h] + (size_t)(i - 1) * DEMB * DEMB, DEMB, DEMB, DEMB);
  }

  // --- init aggregation buffers ---
  size_t jc = (size_t)NJ * DEMB;
  float ninf = -__builtin_inff();
  fill_kernel<<<nblk(jc, T), T, 0, stream>>>(sum, 0.f, jc);
  fill_kernel<<<nblk(jc, T), T, 0, stream>>>(sumsq, 0.f, jc);
  fill_kernel<<<nblk(jc, T), T, 0, stream>>>(maxx, ninf, jc);
  fill_kernel<<<nblk(jc, T), T, 0, stream>>>(zmax, ninf, jc);
  fill_kernel<<<nblk(jc, T), T, 0, stream>>>(den, 0.f, jc);
  fill_kernel<<<nblk(jc, T), T, 0, stream>>>(num, 0.f, jc);
  fill_kernel<<<nblk((size_t)NJ, T), T, 0, stream>>>(cnt, 0.f, (size_t)NJ);

  // --- embedding FFN over particles ---
  EmbArgs ea;
  ea.pf = pf; ea.W0t = embW0t;
  for (int i = 0; i < 5; ++i) ea.Wt[i] = embWt + (size_t)i * DEMB * DEMB;
  for (int i = 0; i < 6; ++i) ea.b[i] = emb.b[i];
  for (int i = 0; i < 5; ++i) { ea.g[i] = emb.g[i]; ea.be[i] = emb.be[i]; }
  ea.out = pf_enc;
  emb_ffn_kernel<<<NPF / MT, T, 0, stream>>>(ea);

  // --- aggregations ---
  size_t nc = (size_t)NPF * DEMB;
  agg_pass1<<<nblk(nc, T), T, 0, stream>>>(pf_enc, idx, tptr, sum, sumsq, maxx, zmax, cnt);
  agg_pass2<<<nblk(nc, T), T, 0, stream>>>(pf_enc, idx, tptr, zmax, den, num);
  finalize_kernel<<<NJ, T, 0, stream>>>(jetf, sum, sumsq, maxx, den, num, cnt, feats);

  // --- heads ---
  float* out_istau = (float*)d_out;                // [NJ][2]
  float* out_p4 = (float*)d_out + (size_t)NJ * 2;  // [NJ][4]
  for (int h = 0; h < 2; ++h) {
    HeadArgs ha;
    ha.feats = feats; ha.W0t = hW0t[h];
    for (int i = 0; i < 4; ++i) ha.Wt[i] = hWt[h] + (size_t)i * DEMB * DEMB;
    for (int i = 0; i < 5; ++i) { ha.b[i] = heads[h]->b[i]; ha.g[i] = heads[h]->g[i]; ha.be[i] = heads[h]->be[i]; }
    ha.W5 = heads[h]->W[5]; ha.b5 = heads[h]->b[5];
    ha.jetf = (h == 1) ? jetf : nullptr;
    ha.out = (h == 0) ? out_istau : out_p4;
    ha.nout = (h == 0) ? 2 : 4;
    head_ffn_kernel<<<NJ / MT, T, 0, stream>>>(ha);
  }
  (void)n_in; (void)out_size; (void)ws_size;
}
